// GraphAttentionReadOut_80728205296198
// MI455X (gfx1250) — compile-verified, matrix operand order checked by reading, not run on hardware
//
#include <hip/hip_runtime.h>
#include <cmath>

typedef __attribute__((ext_vector_type(16))) _Float16 v16h;
typedef __attribute__((ext_vector_type(8)))  float    v8f;

#define DFEA 64
#define NHEAD 3
#define HID 32

// 16-bit WMMA operand K-index mapping (per CDNA5 ISA 7.12.2):
// lanes 0-15: elements 0..7 -> K=0..7,  elements 8..15 -> K=16..23
// lanes 16-31: elements 0..7 -> K=8..15, elements 8..15 -> K=24..31
__device__ __forceinline__ int kmap(int laneHi, int e) {
    return (e & 7) + ((e >> 3) << 4) + (laneHi << 3);
}

__device__ __forceinline__ int lowerBound(const int* __restrict__ a, int n, int key) {
    int lo = 0, hi = n;
    while (lo < hi) {
        int mid = (lo + hi) >> 1;
        if (a[mid] < key) lo = mid + 1; else hi = mid;
    }
    return lo;
}

__device__ __forceinline__ v8f wmma_f16(v16h a, v16h b, v8f c) {
    return __builtin_amdgcn_wmma_f32_16x16x32_f16(
        /*neg_a=*/false, a, /*neg_b=*/false, b,
        /*c_mod=*/(short)0, c, /*reuse_a=*/false, /*reuse_b=*/false);
}

// fast SiLU: x * rcp(1 + exp(-x))  (v_exp_f32 + v_rcp_f32, no IEEE div chain)
__device__ __forceinline__ float fast_silu(float x) {
    return x * __builtin_amdgcn_rcpf(1.0f + __expf(-x));
}

__global__ __launch_bounds__(32)
void gar_fused_kernel(const float* __restrict__ feas,
                      const float* __restrict__ w1,
                      const float* __restrict__ b1,
                      const float* __restrict__ w2,
                      const float* __restrict__ b2,
                      const int*   __restrict__ owner,
                      float* __restrict__ out,
                      int n_atoms, int n_seg) {
    __shared__ float    ldsF[16][DFEA];   // chunk features, f32 (4 KB)
    __shared__ _Float16 ldsH[16][HID];    // hidden activations, f16 (1 KB)
    __shared__ float    ldsS[16][4];      // per-atom head scores (pad to 4)

    const int g = blockIdx.x;
    if (g >= n_seg) return;
    const int lane   = threadIdx.x;     // wave32
    const int laneLo = lane & 15;
    const int laneHi = lane >> 4;

    // contiguous segment bounds (owner is sorted); uniform across the wave
    const int segStart = lowerBound(owner, n_atoms, g);
    const int segEnd   = lowerBound(owner, n_atoms, g + 1);

    float* outRow = out + (size_t)g * (DFEA * NHEAD);
    if (segStart >= segEnd) {            // empty segment -> zeros (segment_sum identity)
        for (int i = lane; i < DFEA * NHEAD; i += 32) outRow[i] = 0.0f;
        return;
    }

    // ---- Pack B operands once per wave (registers, reused every chunk) ----
    // W1: [64 (K) x 32 (N)] row-major -> 2 K-blocks x 2 N-halves
    v16h bw1[2][2];
    #pragma unroll
    for (int kb = 0; kb < 2; ++kb)
        #pragma unroll
        for (int nb = 0; nb < 2; ++nb)
            #pragma unroll
            for (int e = 0; e < 16; ++e) {
                int K = kb * 32 + kmap(laneHi, e);
                int Nc = nb * 16 + laneLo;
                bw1[kb][nb][e] = (_Float16)w1[K * HID + Nc];
            }
    // W2: [32 (K) x 3 (N)] padded to 32x16
    v16h bw2;
    #pragma unroll
    for (int e = 0; e < 16; ++e) {
        int K = kmap(laneHi, e);
        bw2[e] = (laneLo < NHEAD) ? (_Float16)w2[K * NHEAD + laneLo] : (_Float16)0.0f;
    }
    const float b1lo = b1[laneLo];
    const float b1hi = b1[16 + laneLo];
    const float b2v  = (laneLo < NHEAD) ? b2[laneLo] : 0.0f;

    // ---- Online-softmax running state ----
    float m[NHEAD], denom[NHEAD], acc[NHEAD][2];
    #pragma unroll
    for (int h = 0; h < NHEAD; ++h) {
        m[h] = -3.0e38f; denom[h] = 0.0f; acc[h][0] = 0.0f; acc[h][1] = 0.0f;
    }
    // lane owns feature columns d=lane and d=lane+32 (conflict-free LDS reads)
    const int dA = lane;
    const int dB = lane + 32;

    for (int base = segStart; base < segEnd; base += 16) {
        const int cnt = min(16, segEnd - base);

        // prefetch next chunk (global_prefetch_b8), one cacheline per lane
        if (base + 16 < segEnd) {
            const char* np = (const char*)(feas + (size_t)(base + 16) * DFEA);
            __builtin_prefetch(np + lane * 128, 0, 0);
        }

        // ---- Stage chunk features in LDS (coalesced 128B per row-half) ----
        #pragma unroll 4
        for (int r = 0; r < 16; ++r) {
            float v0 = 0.0f, v1 = 0.0f;
            if (r < cnt) {
                const float* p = feas + (size_t)(base + r) * DFEA;
                v0 = p[lane]; v1 = p[lane + 32];
            }
            ldsF[r][lane] = v0; ldsF[r][lane + 32] = v1;
        }
        __syncthreads();

        // ---- Layer 1: [16x64] @ [64x32] via 4 WMMAs ----
        v16h aK0, aK1;
        #pragma unroll
        for (int e = 0; e < 16; ++e) {
            int k = kmap(laneHi, e);
            aK0[e] = (_Float16)ldsF[laneLo][k];
            aK1[e] = (_Float16)ldsF[laneLo][32 + k];
        }
        v8f h0 = {}; v8f h1 = {};
        h0 = wmma_f16(aK0, bw1[0][0], h0);
        h0 = wmma_f16(aK1, bw1[1][0], h0);
        h1 = wmma_f16(aK0, bw1[0][1], h1);
        h1 = wmma_f16(aK1, bw1[1][1], h1);

        // bias + SiLU, write hidden (f16) to LDS in row-major for repacking
        #pragma unroll
        for (int r = 0; r < 8; ++r) {
            int row = r + laneHi * 8;
            float x = fast_silu(h0[r] + b1lo);
            float y = fast_silu(h1[r] + b1hi);
            ldsH[row][laneLo]      = (_Float16)x;
            ldsH[row][16 + laneLo] = (_Float16)y;
        }
        __syncthreads();

        // ---- Layer 2: [16x32] @ [32x16(3 valid)] via 1 WMMA ----
        v16h a2;
        #pragma unroll
        for (int e = 0; e < 16; ++e) a2[e] = ldsH[laneLo][kmap(laneHi, e)];
        v8f sc = {};
        sc = wmma_f16(a2, bw2, sc);

        // scatter scores (+b2) to LDS: lane holds column N=laneLo
        if (laneLo < NHEAD) {
            #pragma unroll
            for (int r = 0; r < 8; ++r) ldsS[r + laneHi * 8][laneLo] = sc[r] + b2v;
        }
        __syncthreads();

        // ---- Online softmax update + weighted pooling ----
        #pragma unroll
        for (int h = 0; h < NHEAD; ++h) {
            float cmax = -3.0e38f;
            for (int a = 0; a < cnt; ++a) cmax = fmaxf(cmax, ldsS[a][h]);
            float mNew  = fmaxf(m[h], cmax);
            float scale = __expf(m[h] - mNew);
            denom[h] *= scale; acc[h][0] *= scale; acc[h][1] *= scale;
            for (int a = 0; a < cnt; ++a) {
                float w = __expf(ldsS[a][h] - mNew);
                denom[h] += w;
                acc[h][0] = fmaf(w, ldsF[a][dA], acc[h][0]);
                acc[h][1] = fmaf(w, ldsF[a][dB], acc[h][1]);
            }
            m[h] = mNew;
        }
        __syncthreads();   // before next chunk overwrites ldsF/ldsS
    }

    // ---- out[g, d, h], head fastest (torch .view(-1) order) ----
    #pragma unroll
    for (int h = 0; h < NHEAD; ++h) {
        float r = __builtin_amdgcn_rcpf(denom[h]);
        outRow[dA * NHEAD + h] = acc[h][0] * r;
        outRow[dB * NHEAD + h] = acc[h][1] * r;
    }
}

extern "C" void kernel_launch(void* const* d_in, const int* in_sizes, int n_in,
                              void* d_out, int out_size, void* d_ws, size_t ws_size,
                              hipStream_t stream) {
    const float* feas  = (const float*)d_in[0];   // [N, 64]
    const float* w1    = (const float*)d_in[1];   // [64, 32]
    const float* b1    = (const float*)d_in[2];   // [32]
    const float* w2    = (const float*)d_in[3];   // [32, 3]
    const float* b2    = (const float*)d_in[4];   // [3]
    const int*   owner = (const int*)d_in[5];     // [N] sorted
    float* out = (float*)d_out;                   // [G, 192]

    const int n_atoms = in_sizes[0] / DFEA;
    const int n_seg   = out_size / (DFEA * NHEAD);   // G (device scalar unreadable in capture)

    gar_fused_kernel<<<n_seg, 32, 0, stream>>>(feas, w1, b1, w2, b2, owner,
                                               out, n_atoms, n_seg);
}
